// SelfAttention_86105504350805
// MI455X (gfx1250) — matrix operand order repaired and checked by reading, not compiled, not run on hardware
//
#include <hip/hip_runtime.h>
#include <hip/hip_bf16.h>

// ---------------------------------------------------------------------------
// Non-local (gaussian) attention block, MI455X / gfx1250, bf16 WMMA pipeline.
//   B=4, C=256, H=W=64, N=4096
//   out = x + gamma * w_out @ softmax(theta^T phi) applied to g
//
// Pipeline:
//   k_prep : fp32 -> bf16 one-shot conversion of the four 256x256 weights
//   k_proj : theta/phi (token-major) and g (channel-major) projections, WMMA
//   k_attn : fused flash attention per 16-query tile, WMMA + online softmax
//   k_out  : output projection + residual, WMMA
// ---------------------------------------------------------------------------

#define BATCH 4
#define CH    256
#define NTOK  4096

typedef __bf16 bf16_t;
typedef __attribute__((ext_vector_type(16))) __bf16 v16bf;
typedef __attribute__((ext_vector_type(8)))  __bf16 v8bf;
typedef __attribute__((ext_vector_type(8)))  float  v8f;
typedef __attribute__((ext_vector_type(4)))  float  v4f;

// D = A(16x32 bf16) * B(32x16 bf16) + C(16x16 f32)
__device__ inline v8f wmma_bf16(v16bf a, v16bf b, v8f c) {
  return __builtin_amdgcn_wmma_f32_16x16x32_bf16(
      /*neg_a=*/false, a, /*neg_b=*/false, b,
      /*c_mod=*/(short)0, c, /*reuse_a=*/false, /*reuse_b=*/false);
}

// A-fragment gather for a 16x32 bf16 tile from a K-contiguous row (row*C + k0
// already folded into p).  Lanes 0-15 hold K {0..15},
// lanes 16-31 hold K {16..31}.  Two contiguous 16B loads per lane.
__device__ inline v16bf load_afrag(const bf16_t* p, int lhalf) {
  const bf16_t* q = p + (lhalf << 4);
  v8bf lo = *(const v8bf*)q;          // K = kb .. kb+7
  v8bf hi = *(const v8bf*)(q + 8);    // K = kb+8 .. kb+15
  return __builtin_shufflevector(lo, hi, 0,1,2,3,4,5,6,7,8,9,10,11,12,13,14,15);
}

// ---------------------------------------------------------------------------
// Kernel 0: one-shot weight conversion fp32 -> bf16 (4 matrices, 1MB in).
// Layout preserved ([o][c] row-major) so B-fragments are contiguous.
// Grid: 256 blocks x 256 threads, 4 elements per thread.
// ---------------------------------------------------------------------------
__global__ __launch_bounds__(256) void k_prep(
    const float* __restrict__ wt, const float* __restrict__ wp,
    const float* __restrict__ wg, const float* __restrict__ wo,
    bf16_t* __restrict__ wbf)   // [4][CH*CH]
{
  const int idx = (blockIdx.x * 256 + threadIdx.x) * 4;   // 0 .. 262140
  const int m   = idx >> 16;                              // matrix id
  const int off = idx & 65535;
  const float* src = (m == 0) ? wt : (m == 1) ? wp : (m == 2) ? wg : wo;
  v4f f = *(const v4f*)(src + off);
  bf16_t* dst = wbf + (m << 16) + off;
  dst[0] = (bf16_t)f.x; dst[1] = (bf16_t)f.y;
  dst[2] = (bf16_t)f.z; dst[3] = (bf16_t)f.w;
}

// ---------------------------------------------------------------------------
// Kernel 1: projections.  theta/phi stored token-major [b][n][c] (bf16),
// g stored channel-major [b][c][n] (bf16).
// Grid: BATCH * (NTOK/64) blocks of 256 threads (8 waves).
// ---------------------------------------------------------------------------
__global__ __launch_bounds__(256) void k_proj(
    const float* __restrict__ x,
    const bf16_t* __restrict__ wbf,   // [3 used][CH*CH] bf16
    bf16_t* __restrict__ theta, bf16_t* __restrict__ phi,
    bf16_t* __restrict__ g)
{
  __shared__ bf16_t xt[64 * CH];              // x tile, token-major [t][c], 32KB
  const int b  = blockIdx.x >> 6;
  const int n0 = (blockIdx.x & 63) << 6;
  const int tid = threadIdx.x;

  // Stage x tile (coalesced reads: 64 consecutive tokens per channel).
  for (int i = 0; i < 64; ++i) {
    int idx = i * 256 + tid;                  // 0..16383
    int c = idx >> 6, t = idx & 63;
    float v = x[(size_t)(b * CH + c) * NTOK + n0 + t];
    xt[t * CH + c] = (bf16_t)v;
  }
  __syncthreads();

  const int lane = tid & 31, wave = tid >> 5;
  const int lhalf = lane >> 4, l15 = lane & 15;

  // 3 projections x 4 n-subtiles x 16 o-subtiles = 192 WMMA output tiles.
  for (int work = wave; work < 192; work += 8) {
    const int p = work / 64;
    const int t = work % 64;
    const int nsub = t >> 4, osub = t & 15;

    v8f acc = {};
    const bf16_t* arow = &xt[((nsub << 4) + l15) * CH];
    // B fragment: lane col o = osub*16+l15, K = k0 + e + 16*lhalf -> one
    // contiguous 32B bf16 load per K-step from the pre-converted weights.
    const bf16_t* wrow =
        wbf + ((size_t)p << 16) + (size_t)((osub << 4) + l15) * CH + (lhalf << 4);
#pragma unroll
    for (int kk = 0; kk < 8; ++kk) {
      v16bf a  = load_afrag(arow + (kk << 5), lhalf);
      v16bf bf = *(const v16bf*)(wrow + (kk << 5));
      acc = wmma_bf16(a, bf, acc);
    }

    const int o = (osub << 4) + l15;
    if (p < 2) {
      // token-major scatter (row stride C)
      bf16_t* dst = (p == 0) ? theta : phi;
#pragma unroll
      for (int v = 0; v < 8; ++v) {
        int n = n0 + (nsub << 4) + (lhalf << 3) + v;
        dst[(size_t)(b * NTOK + n) * CH + o] = (bf16_t)acc[v];
      }
    } else {
      // channel-major: 8 consecutive tokens per lane -> one 16B store
      v8bf pk;
#pragma unroll
      for (int v = 0; v < 8; ++v) pk[v] = (bf16_t)acc[v];
      int n = n0 + (nsub << 4) + (lhalf << 3);
      *(v8bf*)&g[(size_t)(b * CH + o) * NTOK + n] = pk;
    }
  }
}

// ---------------------------------------------------------------------------
// Kernel 2: fused flash attention.  One block = one batch + 16 query rows.
// 8 waves: wave w scores columns [m0+16w, m0+16w+16) and owns output channels
// [32w, 32w+32).  Grid: BATCH * (NTOK/16) = 1024 blocks of 256 threads.
// ---------------------------------------------------------------------------
__global__ __launch_bounds__(256) void k_attn(
    const bf16_t* __restrict__ theta, const bf16_t* __restrict__ phi,
    const bf16_t* __restrict__ g, bf16_t* __restrict__ y)
{
  __shared__ bf16_t Plds[16 * 128];           // P tile, [row][m_local], 4KB
  __shared__ float  redmax[8][16];
  __shared__ float  redsum[8][16];

  const int b  = blockIdx.x >> 8;
  const int n0 = (blockIdx.x & 255) << 4;
  const int tid = threadIdx.x, lane = tid & 31, wave = tid >> 5;
  const int lhalf = lane >> 4, l15 = lane & 15;

  // Preload theta A-fragments for the full K=256 (8 frags, 64 VGPRs/lane).
  v16bf Afrag[8];
  const bf16_t* trow = theta + (size_t)(b * NTOK + n0 + l15) * CH;
#pragma unroll
  for (int kk = 0; kk < 8; ++kk) Afrag[kk] = load_afrag(trow + (kk << 5), lhalf);

  float run_max[8], run_sum[8];
  v8f yacc0 = {}, yacc1 = {};
#pragma unroll
  for (int v = 0; v < 8; ++v) { run_max[v] = -1e30f; run_sum[v] = 0.f; }

  const int cbase = wave << 4;                // column offset inside 128-chunk
  const int chan0 = wave << 5;                // this wave's 32-channel slice

  for (int m0 = 0; m0 < NTOK; m0 += 128) {
    // ---- scores: S(16x16) = theta(16x256) @ phi_cols(256x16) --------------
    v8f S = {};
    const bf16_t* prow =
        phi + (size_t)(b * NTOK + m0 + cbase + l15) * CH + (lhalf << 4);
    // Warm L2/WGP$ for the next chunk while this chunk computes.
    if (m0 + 128 < NTOK) {
      __builtin_prefetch(prow + (size_t)128 * CH, 0, 0);
      __builtin_prefetch(g + (size_t)(b * CH + chan0 + l15) * NTOK + m0 + 128, 0, 0);
    }
#pragma unroll
    for (int kk = 0; kk < 8; ++kk) {
      v16bf bf = *(const v16bf*)(prow + (kk << 5));
      S = wmma_bf16(Afrag[kk], bf, S);
    }

    // ---- per-wave row max (reduce over 16 lanes of each half) -------------
    float tmax[8];
#pragma unroll
    for (int v = 0; v < 8; ++v) {
      float m = S[v];
      for (int off = 1; off < 16; off <<= 1) m = fmaxf(m, __shfl_xor(m, off, 16));
      tmax[v] = m;
    }
    if (l15 == 0) {
#pragma unroll
      for (int v = 0; v < 8; ++v) redmax[wave][(lhalf << 3) + v] = tmax[v];
    }
    __syncthreads();

    float nmax[8], alpha[8];
#pragma unroll
    for (int v = 0; v < 8; ++v) {
      const int row = (lhalf << 3) + v;
      float m = run_max[v];
#pragma unroll
      for (int w = 0; w < 8; ++w) m = fmaxf(m, redmax[w][row]);
      nmax[v]  = m;
      alpha[v] = __expf(run_max[v] - m);
      run_max[v] = m;
    }

    // ---- P = exp(S - max), partial row sums, stash P in LDS ---------------
    float tsum[8];
#pragma unroll
    for (int v = 0; v < 8; ++v) {
      float pexp = __expf(S[v] - nmax[v]);
      Plds[((lhalf << 3) + v) * 128 + cbase + l15] = (bf16_t)pexp;
      float s = pexp;
      for (int off = 1; off < 16; off <<= 1) s += __shfl_xor(s, off, 16);
      tsum[v] = s;
    }
    if (l15 == 0) {
#pragma unroll
      for (int v = 0; v < 8; ++v) redsum[wave][(lhalf << 3) + v] = tsum[v];
    }
    __syncthreads();

#pragma unroll
    for (int v = 0; v < 8; ++v) {
      const int row = (lhalf << 3) + v;
      float s = 0.f;
#pragma unroll
      for (int w = 0; w < 8; ++w) s += redsum[w][row];
      run_sum[v] = run_sum[v] * alpha[v] + s;
      yacc0[v] *= alpha[v];
      yacc1[v] *= alpha[v];
    }

    // ---- y += P(16x128) @ g_slice(128x32) ---------------------------------
#pragma unroll
    for (int t = 0; t < 2; ++t) {
      const int cch = chan0 + (t << 4) + l15;
      const bf16_t* grow =
          g + (size_t)(b * CH + cch) * NTOK + m0 + (lhalf << 4);
#pragma unroll
      for (int j = 0; j < 4; ++j) {
        const bf16_t* pl = &Plds[l15 * 128 + (j << 5) + (lhalf << 4)];
        v8bf lo = *(const v8bf*)pl;
        v8bf hi = *(const v8bf*)(pl + 8);
        v16bf a = __builtin_shufflevector(lo, hi,
                      0,1,2,3,4,5,6,7,8,9,10,11,12,13,14,15);
        v16bf bf = *(const v16bf*)(grow + (j << 5));
        if (t == 0) yacc0 = wmma_bf16(a, bf, yacc0);
        else        yacc1 = wmma_bf16(a, bf, yacc1);
      }
    }
    __syncthreads();   // protect Plds / reductions before next chunk
  }

  // ---- normalize + store y token-major bf16 -------------------------------
#pragma unroll
  for (int t = 0; t < 2; ++t) {
    const int cch = chan0 + (t << 4) + l15;
#pragma unroll
    for (int v = 0; v < 8; ++v) {
      const int n = n0 + (lhalf << 3) + v;
      float val = ((t == 0) ? yacc0[v] : yacc1[v]) / run_sum[v];
      y[(size_t)(b * NTOK + n) * CH + cch] = (bf16_t)val;
    }
  }
}

// ---------------------------------------------------------------------------
// Kernel 3: att = w_out @ y, out = x + gamma*att (channel-major, contiguous).
// Grid: BATCH * (NTOK/64) blocks of 256 threads.
// ---------------------------------------------------------------------------
__global__ __launch_bounds__(256) void k_out(
    const bf16_t* __restrict__ y, const bf16_t* __restrict__ wo_bf,
    const float* __restrict__ x, const float* __restrict__ gamma,
    float* __restrict__ out)
{
  const int b  = blockIdx.x >> 6;
  const int n0 = (blockIdx.x & 63) << 6;
  const int tid = threadIdx.x, lane = tid & 31, wave = tid >> 5;
  const int lhalf = lane >> 4, l15 = lane & 15;
  const float gm = gamma[0];

  for (int t = wave; t < 64; t += 8) {
    const int nsub = t >> 4, osub = t & 15;
    v8f acc = {};
    const bf16_t* yrow =
        y + (size_t)(b * NTOK + n0 + (nsub << 4) + l15) * CH;
    const bf16_t* wrow =
        wo_bf + (size_t)((osub << 4) + l15) * CH + (lhalf << 4);
#pragma unroll
    for (int kk = 0; kk < 8; ++kk) {
      v16bf a  = load_afrag(yrow + (kk << 5), lhalf);
      v16bf bf = *(const v16bf*)(wrow + (kk << 5));
      acc = wmma_bf16(a, bf, acc);
    }
    const int o = (osub << 4) + l15;
    const size_t base =
        (size_t)(b * CH + o) * NTOK + n0 + (nsub << 4) + (lhalf << 3);
#pragma unroll
    for (int v = 0; v < 8; ++v)
      out[base + v] = x[base + v] + gm * acc[v];
  }
}

// ---------------------------------------------------------------------------
extern "C" void kernel_launch(void* const* d_in, const int* in_sizes, int n_in,
                              void* d_out, int out_size, void* d_ws, size_t ws_size,
                              hipStream_t stream) {
  const float* x  = (const float*)d_in[0];
  const float* wt = (const float*)d_in[1];
  const float* wp = (const float*)d_in[2];
  const float* wg = (const float*)d_in[3];
  const float* wo = (const float*)d_in[4];
  const float* gm = (const float*)d_in[5];

  const size_t szProj = (size_t)BATCH * NTOK * CH;   // elements per buffer
  bf16_t* theta = (bf16_t*)d_ws;
  bf16_t* phi   = theta + szProj;
  bf16_t* g     = phi + szProj;
  bf16_t* yws   = g + szProj;
  bf16_t* wbf   = yws + szProj;                      // [4][CH*CH] bf16, 512KB
  // total workspace: 4*8MB + 0.5MB = 32.5MB

  k_prep<<<256, 256, 0, stream>>>(wt, wp, wg, wo, wbf);
  k_proj<<<BATCH * (NTOK / 64), 256, 0, stream>>>(x, wbf, theta, phi, g);
  k_attn<<<BATCH * (NTOK / 16), 256, 0, stream>>>(theta, phi, g, yws);
  k_out <<<BATCH * (NTOK / 64), 256, 0, stream>>>(
      yws, wbf + ((size_t)3 << 16), x, gm, (float*)d_out);
}
